// H_cap_87832081203625
// MI455X (gfx1250) — compile-verified, matrix operand order checked by reading, not run on hardware
//
#include <hip/hip_runtime.h>
#include <math.h>

#define NROWS 8192
#define DDIM  1536

typedef __attribute__((ext_vector_type(16))) __bf16        v16bf;
typedef __attribute__((ext_vector_type(8)))  float         v8f;
typedef __attribute__((ext_vector_type(4)))  float         f4v;
typedef __attribute__((ext_vector_type(4)))  unsigned int  u4v;
typedef __attribute__((ext_vector_type(2)))  unsigned int  u2v;

// fp32 -> bf16 round-to-nearest-even
__device__ __forceinline__ unsigned short f2bf(float x) {
  unsigned int u = __float_as_uint(x);
  u += 0x7FFFu + ((u >> 16) & 1u);
  return (unsigned short)(u >> 16);
}

union FragU { v16bf v; u4v q[2]; };

enum {
  OUT_F32_SCALE = 0,   // C fp32 = acc * alpha          (attention scores)
  OUT_BF16      = 1,   // C bf16 = acc                  (Q1, Q2)
  OUT_BF16_TRANS= 2,   // C bf16 transposed             (Rt)
  OUT_BF16_ADD  = 3,   // C bf16 = acc + E              (X = Zcap + Z)
  OUT_BF16_BRELU= 4,   // C bf16 = relu(acc + E)        (H1)
  OUT_BF16_BIAS = 5,   // C bf16 = acc + E              (H2)
  OUT_F32_BIAS  = 6    // C fp32 = acc + E              (final H)
};

// ---------------------------------------------------------------------------
// bf16 WMMA GEMM: C = epilogue(A[M x K] * B^T)  where
//   A: bf16 row-major M x K (lda), B: bf16 row-major Ncol x K (ldb).
// Block tile 128x128, BK=64, 8 waves (4 along M x 2 along N), wave tile 32x64.
// Staging uses CDNA5 GLOBAL_LOAD_ASYNC_TO_LDS_B128 (ASYNCcnt) with
// double-buffered LDS: async loads for tile k+1 overlap the 16 WMMAs of tile k.
// Requires K % 128 == 0 (even number of 64-wide K tiles) -- true for 1536/8192.
// ---------------------------------------------------------------------------
template <int OUT>
__global__ __launch_bounds__(256)
void gemm_bf16_wmma(const unsigned short* __restrict__ A,
                    const unsigned short* __restrict__ B,
                    const float* __restrict__ E, void* __restrict__ C,
                    int K, long lda, long ldb, long ldc, float alpha)
{
  constexpr int LDT = 72;                       // LDS row stride (bf16): 144B, conflict-free b128
  __shared__ unsigned short sa0[128 * LDT];     // A tile buf0 [m][k]
  __shared__ unsigned short sb0[128 * LDT];     // B tile buf0 [n][k]
  __shared__ unsigned short sa1[128 * LDT];     // A tile buf1
  __shared__ unsigned short sb1[128 * LDT];     // B tile buf1

  const int tid  = threadIdx.x;
  const int lane = tid & 31;
  const int wave = tid >> 5;
  const int wm0  = (wave & 3) * 32;
  const int wn0  = (wave >> 2) * 64;
  const int bm   = blockIdx.y * 128;
  const int bn   = blockIdx.x * 128;

  const int fr = lane & 15;                     // row(A)/col(B) within 16x16 tile
  const int kb = (lane >> 4) * 8;               // K sub-block per ISA 16-bit layout

  // staging geometry: 1024 b128 chunks per operand tile, 4 per thread
  const int sm = tid >> 3;                      // row 0..31 (+32 per pass)
  const int k8 = tid & 7;                       // 8-element k chunk

  // per-lane LDS byte addresses (low 32 bits of generic LDS pointer = DS address)
  const unsigned ldsOfs = (unsigned)((sm * LDT + k8 * 8) * 2);
  const unsigned lA0 = (unsigned)(uintptr_t)(void*)sa0 + ldsOfs;
  const unsigned lB0 = (unsigned)(uintptr_t)(void*)sb0 + ldsOfs;
  const unsigned lA1 = (unsigned)(uintptr_t)(void*)sa1 + ldsOfs;
  const unsigned lB1 = (unsigned)(uintptr_t)(void*)sb1 + ldsOfs;

  // per-lane global byte offsets (GVS: saddr = matrix base, vaddr = 32-bit offset)
  const unsigned gaOff   = (unsigned)(((size_t)(bm + sm) * lda + k8 * 8) * 2);
  const unsigned gbOff   = (unsigned)(((size_t)(bn + sm) * ldb + k8 * 8) * 2);
  const unsigned gaRow32 = (unsigned)(lda * 64);   // 32 rows * 2B
  const unsigned gbRow32 = (unsigned)(ldb * 64);

  v8f acc[2][4];
  #pragma unroll
  for (int i = 0; i < 2; ++i)
    #pragma unroll
    for (int j = 0; j < 4; ++j) acc[i][j] = v8f{};

  auto issueAsync = [&](unsigned la, unsigned lb, int k0) {
    const unsigned kByte = (unsigned)k0 * 2u;
    #pragma unroll
    for (int it = 0; it < 4; ++it) {
      asm volatile("global_load_async_to_lds_b128 %0, %1, %2"
                   :: "v"(la + (unsigned)(it * (32 * LDT * 2))),
                      "v"(gaOff + kByte + (unsigned)it * gaRow32),
                      "s"(A)
                   : "memory");
      asm volatile("global_load_async_to_lds_b128 %0, %1, %2"
                   :: "v"(lb + (unsigned)(it * (32 * LDT * 2))),
                      "v"(gbOff + kByte + (unsigned)it * gbRow32),
                      "s"(B)
                   : "memory");
    }
  };

  auto computeTile = [&](const unsigned short* ta, const unsigned short* tb) {
    #pragma unroll
    for (int s = 0; s < 2; ++s) {               // two k32 steps per 64-wide tile
      FragU a[2], b[4];
      #pragma unroll
      for (int mt = 0; mt < 2; ++mt) {
        const unsigned short* p = ta + (wm0 + mt * 16 + fr) * LDT + s * 32 + kb;
        a[mt].q[0] = *(const u4v*)(p);
        a[mt].q[1] = *(const u4v*)(p + 16);
      }
      #pragma unroll
      for (int nt = 0; nt < 4; ++nt) {
        const unsigned short* p = tb + (wn0 + nt * 16 + fr) * LDT + s * 32 + kb;
        b[nt].q[0] = *(const u4v*)(p);
        b[nt].q[1] = *(const u4v*)(p + 16);
      }
      #pragma unroll
      for (int mt = 0; mt < 2; ++mt)
        #pragma unroll
        for (int nt = 0; nt < 4; ++nt)
          acc[mt][nt] = __builtin_amdgcn_wmma_f32_16x16x32_bf16(
              false, a[mt].v, false, b[nt].v, (short)0, acc[mt][nt], false, false);
    }
  };

  const int kIters = K / 64;                    // even by construction
  issueAsync(lA0, lB0, 0);
  for (int kk = 0; kk < kIters; kk += 2) {
    asm volatile("s_wait_asynccnt 0x0" ::: "memory");
    __syncthreads();                            // buf0 visible from all waves; buf1 readers done
    if (kk + 1 < kIters) issueAsync(lA1, lB1, (kk + 1) * 64);
    computeTile(sa0, sb0);

    asm volatile("s_wait_asynccnt 0x0" ::: "memory");
    __syncthreads();                            // buf1 visible; buf0 readers done
    if (kk + 2 < kIters) issueAsync(lA0, lB0, (kk + 2) * 64);
    computeTile(sa1, sb1);
  }

  // ---- epilogue: C/D tile layout: lane%16 = N, lane/16 -> M 0-7 / 8-15, vgpr = M%8 ----
  const int r0 = (lane >> 4) * 8;
  const int cn = lane & 15;
  float*          Cf = (float*)C;
  unsigned short* Cb = (unsigned short*)C;
  #pragma unroll
  for (int mt = 0; mt < 2; ++mt) {
    #pragma unroll
    for (int nt = 0; nt < 4; ++nt) {
      const int gm = bm + wm0 + mt * 16 + r0;
      const int gn = bn + wn0 + nt * 16 + cn;
      if constexpr (OUT == OUT_BF16_TRANS) {
        u4v w;
        #pragma unroll
        for (int j = 0; j < 4; ++j) {
          w[j] = (unsigned)f2bf(acc[mt][nt][2 * j]) |
                 ((unsigned)f2bf(acc[mt][nt][2 * j + 1]) << 16);
        }
        *(u4v*)(Cb + (size_t)gn * ldc + gm) = w;
      } else {
        #pragma unroll
        for (int v = 0; v < 8; ++v) {
          size_t off = (size_t)(gm + v) * ldc + gn;
          float val = acc[mt][nt][v];
          if constexpr (OUT == OUT_F32_SCALE) { Cf[off] = val * alpha; }
          if constexpr (OUT == OUT_BF16)      { Cb[off] = f2bf(val); }
          if constexpr (OUT == OUT_BF16_ADD)  { Cb[off] = f2bf(val + E[off]); }
          if constexpr (OUT == OUT_BF16_BRELU){ Cb[off] = f2bf(fmaxf(val + E[off], 0.0f)); }
          if constexpr (OUT == OUT_BF16_BIAS) { Cb[off] = f2bf(val + E[off]); }
          if constexpr (OUT == OUT_F32_BIAS)  { Cf[off] = val + E[off]; }
        }
      }
    }
  }
}

// fp32 -> bf16 elementwise (vectorized), n4 = count/4
__global__ __launch_bounds__(256)
void cvt_f32_bf16(const float* __restrict__ src, unsigned short* __restrict__ dst, long n4)
{
  long i = (long)blockIdx.x * 256 + threadIdx.x;
  long stride = (long)gridDim.x * 256;
  for (; i < n4; i += stride) {
    f4v v = *(const f4v*)(src + i * 4);
    u2v p;
    p.x = (unsigned)f2bf(v.x) | ((unsigned)f2bf(v.y) << 16);
    p.y = (unsigned)f2bf(v.z) | ((unsigned)f2bf(v.w) << 16);
    *(u2v*)(dst + i * 4) = p;
  }
}

// dst[c][r] (bf16, ld=rows) = src[r][c] (fp32, ld=cols): 32x32 LDS tiles
__global__ __launch_bounds__(256)
void transcvt_f32_bf16(const float* __restrict__ src, unsigned short* __restrict__ dst,
                       int rows, int cols)
{
  __shared__ float t[32][33];
  const int tx = threadIdx.x & 31, ty = threadIdx.x >> 5;
  const int r0 = blockIdx.y * 32, c0 = blockIdx.x * 32;
  #pragma unroll
  for (int i = 0; i < 4; ++i) {
    int r = ty + i * 8;
    t[r][tx] = src[(size_t)(r0 + r) * cols + (c0 + tx)];
  }
  __syncthreads();
  #pragma unroll
  for (int i = 0; i < 4; ++i) {
    int c = ty + i * 8;
    dst[(size_t)(c0 + c) * rows + (r0 + tx)] = f2bf(t[tx][c]);
  }
}

// In-place row softmax on fp32 S; also emits bf16 shadow Sb for the S@R GEMM.
__global__ __launch_bounds__(256)
void softmax_rows(float* __restrict__ S, unsigned short* __restrict__ Sb, int n)
{
  __shared__ float row[NROWS];
  __shared__ float red[256];
  const int tid = threadIdx.x;
  const size_t base = (size_t)blockIdx.x * n;

  float m = -3.402823466e+38f;
  for (int i = tid; i < n; i += 256) { float x = S[base + i]; row[i] = x; m = fmaxf(m, x); }
  red[tid] = m; __syncthreads();
  for (int s = 128; s > 0; s >>= 1) { if (tid < s) red[tid] = fmaxf(red[tid], red[tid + s]); __syncthreads(); }
  m = red[0]; __syncthreads();

  float sum = 0.0f;
  for (int i = tid; i < n; i += 256) { float e = __expf(row[i] - m); row[i] = e; sum += e; }
  red[tid] = sum; __syncthreads();
  for (int s = 128; s > 0; s >>= 1) { if (tid < s) red[tid] += red[tid + s]; __syncthreads(); }
  const float inv = 1.0f / red[0];
  for (int i = tid; i < n; i += 256) {
    float o = row[i] * inv;
    S[base + i]  = o;
    Sb[base + i] = f2bf(o);
  }
}

extern "C" void kernel_launch(void* const* d_in, const int* in_sizes, int n_in,
                              void* d_out, int out_size, void* d_ws, size_t ws_size,
                              hipStream_t stream)
{
  (void)in_sizes; (void)n_in; (void)out_size; (void)ws_size;
  const int    N  = NROWS, D = DDIM;
  const size_t ND = (size_t)N * D;
  const size_t NN = (size_t)N * N;
  const size_t DD = (size_t)D * D;

  const float* Z   = (const float*)d_in[0];
  const float* W_f[6] = { (const float*)d_in[1], (const float*)d_in[2], (const float*)d_in[3],
                          (const float*)d_in[4], (const float*)d_in[5], (const float*)d_in[6] };
  const float* b1  = (const float*)d_in[7];
  const float* b2  = (const float*)d_in[8];
  const float* b3  = (const float*)d_in[9];

  float* out  = (float*)d_out;
  float* Hout = out;                 // [N, D]
  float* S    = out + ND;            // [N, N]
  float* Zout = out + ND + NN;       // [N, D]

  // bf16 workspace (peak ~263 MB)
  unsigned short* wsh = (unsigned short*)d_ws;
  unsigned short* Zb  = wsh;                       // [N, D]
  unsigned short* Wt[6];
  for (int i = 0; i < 6; ++i) Wt[i] = wsh + ND + (size_t)i * DD;   // [D, D] transposed
  unsigned short* Rt  = wsh + ND + 6 * DD;         // [D, N]  (R transposed)
  unsigned short* Q1b = Rt  + ND;                  // [N, D]
  unsigned short* Q2b = Q1b + ND;                  // [N, D]
  unsigned short* Sb  = Q2b + ND;                  // [N, N]
  unsigned short* Xb  = Q1b;                       // reuse (Q1 dead after S)
  unsigned short* H1b = Q2b;                       // reuse (Q2 dead after S)
  unsigned short* H2b = Rt;                        // reuse (Rt dead after Z_cap)

  dim3 blk(256);
  dim3 gD(D / 128, N / 128);         // 12 x 64
  dim3 gS(N / 128, N / 128);         // 64 x 64
  dim3 gT(D / 32, D / 32);           // weight transpose tiles
  const float scale = 1.0f / sqrtf((float)D);

  // One-time conversions (bandwidth-trivial)
  cvt_f32_bf16<<<4096, blk, 0, stream>>>(Z, Zb, (long)(ND / 4));
  for (int i = 0; i < 6; ++i)
    transcvt_f32_bf16<<<gT, blk, 0, stream>>>(W_f[i], Wt[i], D, D);

  // Projections (R written transposed so S@R stages with pure b128 copies)
  gemm_bf16_wmma<OUT_BF16_TRANS><<<gD, blk, 0, stream>>>(Zb, Wt[0], nullptr, Rt,  D, D, D, (long)N, 1.0f);
  gemm_bf16_wmma<OUT_BF16><<<gD, blk, 0, stream>>>(Zb, Wt[1], nullptr, Q1b, D, D, D, D, 1.0f);
  gemm_bf16_wmma<OUT_BF16><<<gD, blk, 0, stream>>>(Zb, Wt[2], nullptr, Q2b, D, D, D, D, 1.0f);
  // Scores: S = (Q1 @ Q2^T) * scale -> fp32 straight into d_out S slice
  gemm_bf16_wmma<OUT_F32_SCALE><<<gS, blk, 0, stream>>>(Q1b, Q2b, nullptr, S, D, D, D, (long)N, scale);
  // Row softmax (in place) + bf16 shadow
  softmax_rows<<<N, blk, 0, stream>>>(S, Sb, N);
  // X = S @ R + Z  (bf16 out, residual fused)
  gemm_bf16_wmma<OUT_BF16_ADD><<<gD, blk, 0, stream>>>(Sb, Rt, Z, Xb, N, (long)N, (long)N, D, 1.0f);
  // MLP
  gemm_bf16_wmma<OUT_BF16_BRELU><<<gD, blk, 0, stream>>>(Xb,  Wt[3], b1, H1b,  D, D, D, D, 1.0f);
  gemm_bf16_wmma<OUT_BF16_BIAS><<<gD, blk, 0, stream>>>(H1b, Wt[4], b2, H2b,  D, D, D, D, 1.0f);
  gemm_bf16_wmma<OUT_F32_BIAS><<<gD, blk, 0, stream>>>(H2b, Wt[5], b3, Hout, D, D, D, D, 1.0f);
  // Third output: Z passthrough
  hipMemcpyAsync(Zout, Z, ND * sizeof(float), hipMemcpyDeviceToDevice, stream);
}
